// SudokuRNN_73890617360531
// MI455X (gfx1250) — matrix operand order repaired and checked by reading, not compiled, not run on hardware
//
#include <hip/hip_runtime.h>

// SudokuRNN forward on gfx1250: one workgroup (192 threads = 6 waves) per board,
// 81 sequential steps fully LDS-resident, GEMMs via v_wmma_f32_16x16x32_f16.
// All weight B-fragments stay in VGPRs (constant indexing only -> no scratch spill).

typedef _Float16 v16h __attribute__((ext_vector_type(16)));
typedef _Float16 v8h  __attribute__((ext_vector_type(8)));
typedef float    v8f  __attribute__((ext_vector_type(8)));

#define NT 192  // 6 waves: matches the 6 M-tiles of both GEMMs

// 16-bit A-matrix 16x32 fragment (ISA 7.12.2): lane L holds row M=L&15;
// K = kb + e (e<8) / kb + e + 8 (e>=8), kb = (L>=16)?8:0.
__device__ __forceinline__ v16h load_a16(const _Float16* base, int stride, int lane) {
  const int r  = lane & 15;
  const int kb = (lane >> 4) << 3;             // 0 or 8
  const _Float16* p = base + r * stride + kb;  // 16B-aligned by construction
  v8h lo = *reinterpret_cast<const v8h*>(p);       // K = kb+0..kb+7
  v8h hi = *reinterpret_cast<const v8h*>(p + 16);  // K = kb+16..kb+23
  v16h a;
#pragma unroll
  for (int e = 0; e < 8; ++e) { a[e] = lo[e]; a[e + 8] = hi[e]; }
  return a;
}

// 16-bit B-matrix 32x16 fragment: lane L holds column N=L&15;
// K = kb + e (e<8) / kb + e + 8 (e>=8), kb = (L>=16)?8:0.
__device__ __forceinline__ v16h load_b16(const _Float16* base, int stride, int lane) {
  const int n  = lane & 15;
  const int kb = (lane >> 4) << 3;             // 0 or 8
  const _Float16* p = base + kb * stride + n;
  v16h b;
#pragma unroll
  for (int e = 0; e < 16; ++e) b[e] = p[(e < 8 ? e : e + 8) * stride];
  return b;
}

__global__ __launch_bounds__(NT) void sudoku_rnn_kernel(
    const float* __restrict__ gx, const float* __restrict__ gW1,
    const float* __restrict__ gW2, float* __restrict__ gout, int batch) {
  __shared__ float xs[729];          // board one-hot state [81][9]
  __shared__ float xp[729];          // x_pred [81][9]
  __shared__ float us[243];          // unit sums [3][9][9] (row/col/box)
  __shared__ float Y[96 * 16];       // logits (padded)
  __shared__ float score[81];
  __shared__ int   posb[81];
  __shared__ int   emptyf[81];
  __shared__ __attribute__((aligned(16))) _Float16 F[96 * 32];      // features, K-padded
  __shared__ __attribute__((aligned(16))) _Float16 W1T[32 * 112];   // W1^T padded
  __shared__ __attribute__((aligned(16))) _Float16 W2T[128 * 16];   // W2^T padded
  __shared__ __attribute__((aligned(16))) _Float16 H[96 * 128];     // hidden, K-padded

  const int tid  = threadIdx.x;
  const int lane = tid & 31;
  const int wave = tid >> 5;          // 0..5
  const int board = blockIdx.x;
  const float* bx = gx + (size_t)board * 729;

  // ---- one-time init: load board, zero-pad matrices, build f16 weight tiles
  for (int idx = tid; idx < 729; idx += NT) { float v = bx[idx]; xs[idx] = v; xp[idx] = v; }
  for (int idx = tid; idx < 96 * 32; idx += NT)  F[idx] = (_Float16)0.f;
  for (int idx = tid; idx < 96 * 128; idx += NT) H[idx] = (_Float16)0.f;
  for (int idx = tid; idx < 32 * 112; idx += NT) {
    int kk = idx / 112, h = idx % 112;
    W1T[kk * 112 + h] = (_Float16)((kk < 27 && h < 100) ? gW1[h * 27 + kk] : 0.f);
  }
  for (int idx = tid; idx < 128 * 16; idx += NT) {
    int hh = idx / 16, v = idx % 16;
    W2T[idx] = (_Float16)((hh < 100 && v < 9) ? gW2[v * 100 + hh] : 0.f);
  }
  __syncthreads();

  // weights are step-invariant: keep B fragments in VGPRs for all 81 steps.
  // NOTE: these arrays are only ever indexed with compile-time constants below,
  // so they remain in registers (no scratch spill).
  v16h w1f[7], w2f[4];
#pragma unroll
  for (int n = 0; n < 7; ++n) w1f[n] = load_b16(&W1T[n * 16], 112, lane);
#pragma unroll
  for (int kc = 0; kc < 4; ++kc) w2f[kc] = load_b16(&W2T[kc * 32 * 16], 16, lane);

  for (int step = 0; step < 81; ++step) {
    // ---- unit sums: us[k][u][v] = sum of x over the 9 cells of unit (k,u)
    for (int t = tid; t < 243; t += NT) {
      int k = t / 81, u = (t % 81) / 9, v = t % 9;
      float s = 0.f;
      if (k == 0) {
#pragma unroll
        for (int j = 0; j < 9; ++j) s += xs[(u * 9 + j) * 9 + v];
      } else if (k == 1) {
#pragma unroll
        for (int j = 0; j < 9; ++j) s += xs[(j * 9 + u) * 9 + v];
      } else {
        int base = (u / 3) * 27 + (u % 3) * 3;
#pragma unroll
        for (int j = 0; j < 9; ++j) s += xs[(base + (j / 3) * 9 + (j % 3)) * 9 + v];
      }
      us[t] = s;
    }
    __syncthreads();

    // ---- scatter to feature matrix F[81][27] (== cmask einsum, exploiting 9-sparsity)
    for (int t = tid; t < 81 * 27; t += NT) {
      int i = t / 27, kk = t % 27;
      int k = kk / 9, v = kk % 9;
      int u = (k == 0) ? (i / 9) : (k == 1) ? (i % 9) : ((i / 27) * 3 + (i % 9) / 3);
      F[i * 32 + kk] = (_Float16)us[k * 81 + u * 9 + v];
    }
    __syncthreads();

    // ---- GEMM1: H[96][112] = relu(F[96][32] @ W1T[32][112])
    // One M-tile per wave; A-fragment loaded once, reused for all 7 N-tiles;
    // w1f indexed by unrolled constant -> stays in VGPRs.
    {
      const int m = wave;  // 0..5
      v16h a = load_a16(&F[m * 16 * 32], 32, lane);
      const int nn = lane & 15, hi8 = (lane >> 4) << 3;  // f32 D tile layout
#pragma unroll
      for (int n = 0; n < 7; ++n) {
        v8f acc = {};
        acc = __builtin_amdgcn_wmma_f32_16x16x32_f16(false, a, false, w1f[n],
                                                     (short)0, acc, false, false);
#pragma unroll
        for (int kr = 0; kr < 8; ++kr) {
          float v = acc[kr];
          H[(m * 16 + hi8 + kr) * 128 + n * 16 + nn] = (_Float16)(v > 0.f ? v : 0.f);
        }
      }
    }
    __syncthreads();

    // ---- GEMM2: Y[96][16] = H[96][128] @ W2T[128][16]; 4 chained WMMAs per M-tile
    {
      const int m = wave;  // 0..5
      v8f acc = {};
#pragma unroll
      for (int kc = 0; kc < 4; ++kc) {
        v16h a = load_a16(&H[(m * 16) * 128 + kc * 32], 128, lane);
        acc = __builtin_amdgcn_wmma_f32_16x16x32_f16(false, a, false, w2f[kc],
                                                     (short)0, acc, false, false);
      }
      const int nn = lane & 15, hi8 = (lane >> 4) << 3;
#pragma unroll
      for (int kr = 0; kr < 8; ++kr)
        Y[(m * 16 + hi8 + kr) * 16 + nn] = acc[kr];
    }
    __syncthreads();

    // ---- softmax + per-cell update (one thread per cell)
    if (tid < 81) {
      float y[9];
      float mx = -3.402823466e38f;
#pragma unroll
      for (int v = 0; v < 9; ++v) { y[v] = Y[tid * 16 + v]; mx = fmaxf(mx, y[v]); }
      float ssum = 0.f;
#pragma unroll
      for (int v = 0; v < 9; ++v) { y[v] = __expf(y[v] - mx); ssum += y[v]; }
      float inv = 1.f / ssum;
      float xsum = 0.f;
#pragma unroll
      for (int v = 0; v < 9; ++v) xsum += xs[tid * 9 + v];
      bool empty = (xsum == 0.f);
      float best = 0.f; int bpos = 0;
#pragma unroll
      for (int v = 0; v < 9; ++v) {
        float p = y[v] * inv;
        if (empty) {
          xp[tid * 9 + v] = p;                  // x_pred = where(empty, probs, x_pred)
          if (p > best) { best = p; bpos = v; } // first-max tie-break like jnp.argmax
        }
      }
      score[tid]  = empty ? best : 0.f;
      posb[tid]   = empty ? bpos : 0;
      emptyf[tid] = empty ? 1 : 0;
    }
    __syncthreads();

    // ---- pick best cell across board (serial scan preserves argmax tie-breaking)
    if (tid == 0) {
      float best = score[0]; int mmax = 0; int any = emptyf[0];
      for (int i = 1; i < 81; ++i) {
        any |= emptyf[i];
        if (score[i] > best) { best = score[i]; mmax = i; }
      }
      if (any) xs[mmax * 9 + posb[mmax]] = 1.f;  // !any => no-op, matches reference
    }
    __syncthreads();
  }

  // ---- outputs: (x_pred, x) concatenated flat
  float* outp = gout + (size_t)board * 729;
  float* outx = gout + (size_t)batch * 729 + (size_t)board * 729;
  for (int idx = tid; idx < 729; idx += NT) { outp[idx] = xp[idx]; outx[idx] = xs[idx]; }
}

extern "C" void kernel_launch(void* const* d_in, const int* in_sizes, int n_in,
                              void* d_out, int out_size, void* d_ws, size_t ws_size,
                              hipStream_t stream) {
  (void)n_in; (void)out_size; (void)d_ws; (void)ws_size;
  const float* x  = (const float*)d_in[0];
  const float* W1 = (const float*)d_in[1];
  const float* W2 = (const float*)d_in[2];
  // d_in[3] (constraint_mask) is a fixed structural mask; row/col/box sums are
  // computed directly from Sudoku geometry, which is mathematically identical.
  int batch = in_sizes[0] / 729;
  sudoku_rnn_kernel<<<batch, NT, 0, stream>>>(x, W1, W2, (float*)d_out, batch);
}
